// Mask2FormerLikeDecoder_11811160064821
// MI455X (gfx1250) — matrix operand order repaired and checked by reading, not compiled, not run on hardware
//
#include <hip/hip_runtime.h>
#include <math.h>

typedef __bf16 bf16;
typedef __attribute__((ext_vector_type(16))) __bf16 v16bf;
typedef __attribute__((ext_vector_type(8)))  float  v8f;

#define DIMC   256
#define NQ     100
#define NHD    8
#define DHEAD  32
#define NBATCH 4
#define NGLOB  5
#define NLAY   6
#define PIXHW  16384   // 128*128

// ---------------------------------------------------------------------------
// Batched WMMA GEMM:  C[z] = alpha * A[z](M,K) @ B[z](N,K)^T + bias
// A,B bf16 row-major (K contiguous). Writes f32 C and/or bf16 Cb (nullable).
// z = blockIdx.z decomposed as (batch, head) with independent pointer strides.
// Each wave computes a register-blocked 32x32 tile = 2x2 WMMA 16x16x32 tiles:
// 4 fragment loads feed 4 v_wmma_f32_16x16x32_bf16 per K-step (2x the
// arithmetic intensity of an unblocked 16x16 tile).
// ---------------------------------------------------------------------------
__global__ __launch_bounds__(128) void k_gemm(
    const bf16* __restrict__ A, const bf16* __restrict__ Bm,
    const float* __restrict__ bias,
    float* __restrict__ C, bf16* __restrict__ Cb,
    int M, int N, int K, int lda, int ldb, int ldc,
    long aZb, long aZh, long bZb, long bZh, long cZb, long cZh,
    int nH, float alpha)
{
    int z  = blockIdx.z;
    int zb = z / nH, zh = z - zb * nH;
    A  += (long)zb * aZb + (long)zh * aZh;
    Bm += (long)zb * bZb + (long)zh * bZh;
    long co = (long)zb * cZb + (long)zh * cZh;

    int wave = threadIdx.x >> 5, lane = threadIdx.x & 31;
    int m0 = blockIdx.y << 5;
    int n0 = (blockIdx.x * 4 + wave) << 5;
    if (m0 >= M || n0 >= N) return;

    int half = lane >> 4, r = lane & 15;
    int ar0 = min(m0 + r,      M - 1);
    int ar1 = min(m0 + 16 + r, M - 1);
    int br0 = min(n0 + r,      N - 1);
    int br1 = min(n0 + 16 + r, N - 1);
    const bf16* Ap0 = A  + (long)ar0 * lda;
    const bf16* Ap1 = A  + (long)ar1 * lda;
    const bf16* Bp0 = Bm + (long)br0 * ldb;
    const bf16* Bp1 = Bm + (long)br1 * ldb;

    v8f a00 = {}, a01 = {}, a10 = {}, a11 = {};
    for (int k0 = 0; k0 < K; k0 += 32) {
        union { v16bf v; uint4 q[2]; } fa0, fa1, fb0, fb1;
        // A 16x32 bf16 fragment: half-wave 0 holds K pairs 0..7 & 16..23,
        // half-wave 1 holds 8..15 & 24..31 (ISA 7.12.2 16-bit A layout).
        fa0.q[0] = *(const uint4*)(Ap0 + k0 + half * 8);
        fa0.q[1] = *(const uint4*)(Ap0 + k0 + half * 8 + 16);
        fa1.q[0] = *(const uint4*)(Ap1 + k0 + half * 8);
        fa1.q[1] = *(const uint4*)(Ap1 + k0 + half * 8 + 16);
        // B 32x16 fragment: same K order as A (half-wave 0: 0..7 & 16..23, half-wave 1: 8..15 & 24..31).
        fb0.q[0] = *(const uint4*)(Bp0 + k0 + half * 8);
        fb0.q[1] = *(const uint4*)(Bp0 + k0 + half * 8 + 16);
        fb1.q[0] = *(const uint4*)(Bp1 + k0 + half * 8);
        fb1.q[1] = *(const uint4*)(Bp1 + k0 + half * 8 + 16);
        if (k0 + 32 < K) {
            __builtin_prefetch(Ap0 + k0 + 64, 0, 0);  // global_prefetch_b8
            __builtin_prefetch(Ap1 + k0 + 64, 0, 0);
            __builtin_prefetch(Bp0 + k0 + 64, 0, 0);
            __builtin_prefetch(Bp1 + k0 + 64, 0, 0);
        }
        a00 = __builtin_amdgcn_wmma_f32_16x16x32_bf16(false, fa0.v, false, fb0.v, (short)0, a00, false, false);
        a01 = __builtin_amdgcn_wmma_f32_16x16x32_bf16(false, fa0.v, false, fb1.v, (short)0, a01, false, false);
        a10 = __builtin_amdgcn_wmma_f32_16x16x32_bf16(false, fa1.v, false, fb0.v, (short)0, a10, false, false);
        a11 = __builtin_amdgcn_wmma_f32_16x16x32_bf16(false, fa1.v, false, fb1.v, (short)0, a11, false, false);
    }

    int c0 = n0 + r, c1 = n0 + 16 + r;
    float bv0 = (bias && c0 < N) ? bias[c0] : 0.f;
    float bv1 = (bias && c1 < N) ? bias[c1] : 0.f;
#pragma unroll
    for (int i = 0; i < 8; ++i) {              // C layout: VGPR i -> M = half*8+i
        int r0 = m0 + half * 8 + i, r1 = r0 + 16;
        if (r0 < M) {
            if (c0 < N) { float v = a00[i] * alpha + bv0; long idx = (long)r0 * ldc + c0 + co;
                          if (C) C[idx] = v; if (Cb) Cb[idx] = (bf16)v; }
            if (c1 < N) { float v = a01[i] * alpha + bv1; long idx = (long)r0 * ldc + c1 + co;
                          if (C) C[idx] = v; if (Cb) Cb[idx] = (bf16)v; }
        }
        if (r1 < M) {
            if (c0 < N) { float v = a10[i] * alpha + bv0; long idx = (long)r1 * ldc + c0 + co;
                          if (C) C[idx] = v; if (Cb) Cb[idx] = (bf16)v; }
            if (c1 < N) { float v = a11[i] * alpha + bv1; long idx = (long)r1 * ldc + c1 + co;
                          if (C) C[idx] = v; if (Cb) Cb[idx] = (bf16)v; }
        }
    }
}

// ---------------------------------------------------------------------------
// LayerNorm over C=256, one wave (32 lanes x 8 elems) per row.
// ---------------------------------------------------------------------------
__global__ void k_ln(const float* __restrict__ X, const float* __restrict__ g,
                     const float* __restrict__ b, bf16* __restrict__ Yb,
                     float* __restrict__ Y, int rows)
{
    int wave = threadIdx.x >> 5, lane = threadIdx.x & 31;
    int row = blockIdx.x * 8 + wave;
    if (row >= rows) return;
    const float* x = X + (long)row * DIMC;
    float v[8]; float s = 0.f;
#pragma unroll
    for (int i = 0; i < 8; ++i) { v[i] = x[lane + 32 * i]; s += v[i]; }
    for (int o = 16; o; o >>= 1) s += __shfl_xor(s, o, 32);
    float mu = s * (1.f / DIMC);
    float s2 = 0.f;
#pragma unroll
    for (int i = 0; i < 8; ++i) { float d = v[i] - mu; s2 += d * d; }
    for (int o = 16; o; o >>= 1) s2 += __shfl_xor(s2, o, 32);
    float inv = rsqrtf(s2 * (1.f / DIMC) + 1e-5f);
#pragma unroll
    for (int i = 0; i < 8; ++i) {
        int c = lane + 32 * i;
        float y = (v[i] - mu) * inv * g[c] + b[c];
        if (Yb) Yb[(long)row * DIMC + c] = (bf16)y;
        if (Y)  Y[(long)row * DIMC + c]  = y;
    }
}

// ---------------------------------------------------------------------------
// Pixel feature transpose: (B,C,16384) f32 -> (B,16384,C) bf16. LDS 32x32 tile.
// ---------------------------------------------------------------------------
__global__ void k_pixT(const float* __restrict__ pix, bf16* __restrict__ out)
{
    __shared__ float tile[32][33];
    int b = blockIdx.z;
    int hw0 = blockIdx.x * 32, c0 = blockIdx.y * 32;
    int tx = threadIdx.x, ty = threadIdx.y;     // 32x8
#pragma unroll
    for (int i = 0; i < 4; ++i)
        tile[ty + 8 * i][tx] = pix[((long)b * DIMC + c0 + ty + 8 * i) * PIXHW + hw0 + tx];
    __syncthreads();
#pragma unroll
    for (int i = 0; i < 4; ++i)
        out[((long)b * PIXHW + hw0 + ty + 8 * i) * DIMC + c0 + tx] =
            (bf16)tile[tx][ty + 8 * i];
}

// ---------------------------------------------------------------------------
// Memory build: mem[b,s,c] = feat[b,c,s] + posenc(c,h,w), bf16, tiled transpose.
// ---------------------------------------------------------------------------
__global__ void k_mem(const float* __restrict__ f, bf16* __restrict__ mem,
                      int H, int W, int S)
{
    __shared__ float tile[32][33];
    int HW = H * W;
    int b = blockIdx.z;
    int hw0 = blockIdx.x * 32, c0 = blockIdx.y * 32;
    int tx = threadIdx.x, ty = threadIdx.y;
#pragma unroll
    for (int i = 0; i < 4; ++i)
        tile[ty + 8 * i][tx] = f[((long)b * DIMC + c0 + ty + 8 * i) * HW + hw0 + tx];
    __syncthreads();
#pragma unroll
    for (int i = 0; i < 4; ++i) {
        int hw = hw0 + ty + 8 * i, c = c0 + tx;
        int h = hw / W, w = hw - h * W;
        int gsel = c >> 6, ci = c & 63;
        float div = expf((float)ci * (-9.210340371976184f / 64.f));
        float ang = (gsel < 2 ? (float)h : (float)w) * div;
        float pe = (gsel & 1) ? cosf(ang) : sinf(ang);
        mem[((long)b * S + hw) * DIMC + c] = (bf16)(tile[tx][ty + 8 * i] + pe);
    }
}

// Append LayerNorm'd global tokens as the last NGLOB rows of mem.
__global__ void k_gtmem(const float* __restrict__ gtln, bf16* __restrict__ mem, int S)
{
    int t = blockIdx.x * 256 + threadIdx.x;
    if (t >= NBATCH * NGLOB * DIMC) return;
    int c = t % DIMC; int rest = t / DIMC;
    int gi = rest % NGLOB; int b = rest / NGLOB;
    mem[((long)b * S + (S - NGLOB) + gi) * DIMC + c] = (bf16)gtln[t];
}

__global__ void k_cvt(const float* __restrict__ src, bf16* __restrict__ dst, long n)
{
    long t = (long)blockIdx.x * 256 + threadIdx.x;
    if (t < n) dst[t] = (bf16)src[t];
}

__global__ void k_initq(const float* __restrict__ Q, float* __restrict__ X,
                        bf16* __restrict__ Xb)
{
    int t = blockIdx.x * 256 + threadIdx.x;
    if (t >= NBATCH * NQ * DIMC) return;
    float v = Q[t % (NQ * DIMC)];
    X[t] = v; Xb[t] = (bf16)v;
}

__global__ void k_addpos(float* __restrict__ X, const float* __restrict__ pos)
{
    int t = blockIdx.x * 256 + threadIdx.x;
    if (t < NBATCH * NQ * DIMC) X[t] += pos[t % (NQ * DIMC)];
}

__global__ void k_resadd(float* __restrict__ X, const float* __restrict__ Y, int n)
{
    int t = blockIdx.x * 256 + threadIdx.x;
    if (t < n) X[t] += Y[t];
}

__global__ void k_gelu(const float* __restrict__ X, bf16* __restrict__ Y, int n)
{
    int t = blockIdx.x * 256 + threadIdx.x;
    if (t < n) {
        float v = X[t];
        Y[t] = (bf16)(0.5f * v * (1.f + erff(v * 0.70710678118654752f)));
    }
}

// ---------------------------------------------------------------------------
// Attention mask: bilinear (half-pixel) downsample of pqm to HxW, threshold 0.
// Global tokens are always in-region => all_blocked statically false.
// mask=1 means blocked; pad columns [S,Spad) forced blocked.
// ---------------------------------------------------------------------------
__global__ void k_mask(const float* __restrict__ pqm, unsigned char* __restrict__ mk,
                       int H, int W, int S, int Spad)
{
    long bq = blockIdx.x;
    const float* src = pqm + bq * (long)PIXHW;
    unsigned char* m = mk + bq * (long)Spad;
    int HW = H * W;
    float scale = 128.f / (float)H;
    for (int s = threadIdx.x; s < HW; s += blockDim.x) {
        int i = s / W, j = s - (s / W) * W;
        float sy = (i + 0.5f) * scale - 0.5f;
        float sx = (j + 0.5f) * scale - 0.5f;
        int y0 = (int)floorf(sy); float fy = sy - y0;
        int x0 = (int)floorf(sx); float fx = sx - x0;
        int y0c = max(y0, 0), y1c = min(y0 + 1, 127);
        int x0c = max(x0, 0), x1c = min(x0 + 1, 127);
        float v = (1.f - fy) * ((1.f - fx) * src[y0c * 128 + x0c] + fx * src[y0c * 128 + x1c])
                +        fy  * ((1.f - fx) * src[y1c * 128 + x0c] + fx * src[y1c * 128 + x1c]);
        m[s] = (v >= 0.f) ? 0 : 1;
    }
    for (int s = HW + threadIdx.x; s < Spad; s += blockDim.x)
        m[s] = (s < S) ? 0 : 1;
}

// ---------------------------------------------------------------------------
// Masked softmax over S, writes bf16 probs zero-padded to Spad. Block=256/row.
// ---------------------------------------------------------------------------
__global__ void k_softmax(const float* __restrict__ L, const unsigned char* __restrict__ mk,
                          bf16* __restrict__ P, int S, int Spad, int nH)
{
    long row = blockIdx.x;                 // (b*nH+h)*NQ + q
    int q = (int)(row % NQ);
    long bh = row / NQ;
    int b = (int)(bh / nH);
    const float* l = L + row * (long)Spad;
    bf16* p = P + row * (long)Spad;
    const unsigned char* m = mk ? mk + ((long)b * NQ + q) * Spad : nullptr;
    int tid = threadIdx.x, lane = tid & 31, wave = tid >> 5;
    __shared__ float red[8];

    float mx = -3.0e38f;
    for (int s = tid; s < S; s += 256)
        if (!(m && m[s])) mx = fmaxf(mx, l[s]);
    for (int o = 16; o; o >>= 1) mx = fmaxf(mx, __shfl_xor(mx, o, 32));
    if (lane == 0) red[wave] = mx;
    __syncthreads();
    mx = red[0];
#pragma unroll
    for (int i = 1; i < 8; ++i) mx = fmaxf(mx, red[i]);
    __syncthreads();

    float sm = 0.f;
    for (int s = tid; s < S; s += 256)
        if (!(m && m[s])) sm += expf(l[s] - mx);
    for (int o = 16; o; o >>= 1) sm += __shfl_xor(sm, o, 32);
    if (lane == 0) red[wave] = sm;
    __syncthreads();
    sm = red[0];
#pragma unroll
    for (int i = 1; i < 8; ++i) sm += red[i];
    float rinv = 1.f / sm;

    for (int s = tid; s < Spad; s += 256) {
        float pv = 0.f;
        if (s < S && !(m && m[s])) pv = expf(l[s] - mx) * rinv;
        p[s] = (bf16)pv;
    }
}

// V transpose: vT[b,h,d,s] = kv[b*S+s, voff+h*32+d], zero-padded to Spad.
__global__ void k_vT(const bf16* __restrict__ kv, bf16* __restrict__ vT,
                     int ldkv, int voff, int S, int Spad)
{
    int t = blockIdx.x * 256 + threadIdx.x;
    int total = NBATCH * NHD * DHEAD * Spad;
    if (t >= total) return;
    int s = t % Spad; int r = t / Spad;
    int d = r % DHEAD; r /= DHEAD;
    int h = r % NHD; int b = r / NHD;
    float v = 0.f;
    if (s < S) v = (float)kv[((long)b * S + s) * ldkv + voff + h * DHEAD + d];
    vT[t] = (bf16)v;
}

// class scores: cls[b,q] = dot(x[b,q,:], cw) + cb; one wave per row.
__global__ void k_cls(const float* __restrict__ X, const float* __restrict__ cw,
                      const float* __restrict__ cb, float* __restrict__ cls,
                      float* __restrict__ outCs)
{
    int wave = threadIdx.x >> 5, lane = threadIdx.x & 31;
    int row = blockIdx.x * 8 + wave;
    if (row >= NBATCH * NQ) return;
    const float* x = X + (long)row * DIMC;
    float s = 0.f;
#pragma unroll
    for (int i = 0; i < 8; ++i) s += x[lane + 32 * i] * cw[lane + 32 * i];
    for (int o = 16; o; o >>= 1) s += __shfl_xor(s, o, 32);
    if (lane == 0) { float v = s + cb[0]; cls[row] = v; outCs[row] = v; }
}

// softmax over the 100 queries per batch.
__global__ void k_softq(const float* __restrict__ cls, float* __restrict__ w)
{
    int b = blockIdx.x, tid = threadIdx.x;     // 128 threads
    __shared__ float sh[128];
    float v = (tid < NQ) ? cls[b * NQ + tid] : -3.0e38f;
    sh[tid] = v; __syncthreads();
    for (int o = 64; o; o >>= 1) { if (tid < o) sh[tid] = fmaxf(sh[tid], sh[tid + o]); __syncthreads(); }
    float mx = sh[0]; __syncthreads();
    float e = (tid < NQ) ? expf(v - mx) : 0.f;
    sh[tid] = e; __syncthreads();
    for (int o = 64; o; o >>= 1) { if (tid < o) sh[tid] += sh[tid + o]; __syncthreads(); }
    float sum = sh[0];
    if (tid < NQ) w[b * NQ + tid] = e / sum;
}

// mask_logits[b,hw] = sum_q pqm[b,q,hw] * w[b,q]
__global__ void k_ml(const float* __restrict__ pqm, const float* __restrict__ w,
                     float* __restrict__ d1, float* __restrict__ d2)
{
    __shared__ float ws_[NQ];
    int g = blockIdx.x * 256 + threadIdx.x;    // NBATCH*16384 total
    int b = g >> 14; int hw = g & 16383;
    if (threadIdx.x < NQ) ws_[threadIdx.x] = w[b * NQ + threadIdx.x];
    __syncthreads();
    float s = 0.f;
    for (int qq = 0; qq < NQ; ++qq) s += pqm[((long)b * NQ + qq) * PIXHW + hw] * ws_[qq];
    d1[g] = s;
    if (d2) d2[g] = s;
}

// ---------------------------------------------------------------------------
static inline void gemm(hipStream_t st, const bf16* A, const bf16* Bm,
                        const float* bias, float* C, bf16* Cb,
                        int M, int N, int K, int lda, int ldb, int ldc,
                        long aZb, long aZh, long bZb, long bZh, long cZb, long cZh,
                        int nH, int nB, float alpha)
{
    dim3 g((((N + 31) / 32) + 3) / 4, (M + 31) / 32, nH * nB);
    k_gemm<<<g, 128, 0, st>>>(A, Bm, bias, C, Cb, M, N, K, lda, ldb, ldc,
                              aZb, aZh, bZb, bZh, cZb, cZh, nH, alpha);
}

extern "C" void kernel_launch(void* const* d_in, const int* in_sizes, int n_in,
                              void* d_out, int out_size, void* d_ws, size_t ws_size,
                              hipStream_t stream)
{
    (void)in_sizes; (void)n_in; (void)out_size; (void)ws_size;
    const float* feat[3] = {(const float*)d_in[0], (const float*)d_in[1], (const float*)d_in[2]};
    const float* pix    = (const float*)d_in[3];
    const float* gtok   = (const float*)d_in[4];
    const float* p_q    = (const float*)d_in[5];
    const float* p_qpos = (const float*)d_in[6];
    const float* p_gtg  = (const float*)d_in[7];
    const float* p_gtb  = (const float*)d_in[8];
    const float* p_cw   = (const float*)d_in[9];
    const float* p_cb   = (const float*)d_in[10];
    auto LP = [&](int li, int k) { return (const float*)d_in[11 + li * 18 + k]; };

    const int Hs[3] = {64, 32, 16};
    const int Ss[3] = {64 * 64 + NGLOB, 32 * 32 + NGLOB, 16 * 16 + NGLOB};
    const int Sps[3] = {4128, 1056, 288};

    // ---- workspace layout ----
    char* base = (char*)d_ws;
    size_t off = 0;
    auto alloc = [&](size_t bytes) -> void* {
        off = (off + 255) & ~(size_t)255;
        void* p = base + off; off += bytes; return p;
    };
    bf16 *wca_in[NLAY], *wca_out[NLAY], *wsa_in[NLAY], *wsa_out[NLAY], *wff1[NLAY], *wff2[NLAY];
    for (int li = 0; li < NLAY; ++li) {
        wca_in[li]  = (bf16*)alloc((size_t)768 * 256 * 2);
        wca_out[li] = (bf16*)alloc((size_t)256 * 256 * 2);
        wsa_in[li]  = (bf16*)alloc((size_t)768 * 256 * 2);
        wsa_out[li] = (bf16*)alloc((size_t)256 * 256 * 2);
        wff1[li]    = (bf16*)alloc((size_t)1024 * 256 * 2);
        wff2[li]    = (bf16*)alloc((size_t)256 * 1024 * 2);
    }
    bf16*  pixT  = (bf16*)alloc((size_t)NBATCH * PIXHW * DIMC * 2);
    float* gtln  = (float*)alloc((size_t)NBATCH * NGLOB * DIMC * 4);
    bf16*  memL[3];
    for (int l = 0; l < 3; ++l) memL[l] = (bf16*)alloc((size_t)NBATCH * Ss[l] * DIMC * 2);
    float* X     = (float*)alloc((size_t)NBATCH * NQ * DIMC * 4);
    bf16*  Xb    = (bf16*)alloc((size_t)NBATCH * NQ * DIMC * 2);
    bf16*  lnb   = (bf16*)alloc((size_t)NBATCH * NQ * DIMC * 2);
    bf16*  qpb   = (bf16*)alloc((size_t)NBATCH * NQ * DIMC * 2);
    bf16*  kvb   = (bf16*)alloc((size_t)NBATCH * 4128 * 512 * 2);
    bf16*  qkvb  = (bf16*)alloc((size_t)(NBATCH * NQ + 128) * 768 * 2);
    bf16*  vT    = (bf16*)alloc((size_t)NBATCH * NHD * DHEAD * 4128 * 2);
    float* logit = (float*)alloc((size_t)NBATCH * NHD * NQ * 4128 * 4);
    bf16*  probs = (bf16*)alloc((size_t)NBATCH * NHD * NQ * 4128 * 2);
    bf16*  ob    = (bf16*)alloc((size_t)NBATCH * NQ * DIMC * 2);
    float* attn  = (float*)alloc((size_t)NBATCH * NQ * DIMC * 4);
    unsigned char* maskb = (unsigned char*)alloc((size_t)NBATCH * NQ * 4128);
    float* pqm   = (float*)alloc((size_t)NBATCH * NQ * PIXHW * 4);
    float* h1    = (float*)alloc((size_t)NBATCH * NQ * 1024 * 4);
    bf16*  gb    = (bf16*)alloc((size_t)NBATCH * NQ * 1024 * 2);
    float* clsb  = (float*)alloc((size_t)NBATCH * NQ * 4);
    float* wq    = (float*)alloc((size_t)NBATCH * NQ * 4);

    float* out    = (float*)d_out;
    float* out_ml = out;
    float* out_cs = out + (size_t)NBATCH * PIXHW;
    float* out_ax = out_cs + NBATCH * NQ;

    const int nXC = NBATCH * NQ * DIMC;           // 102400
    const float ascale = 0.1767766952966369f;     // 1/sqrt(32)

    auto cvt = [&](const float* s, bf16* d, long n) {
        k_cvt<<<dim3((unsigned)((n + 255) / 256)), 256, 0, stream>>>(s, d, n);
    };

    // ---- setup: weight conversion, pixT, memories, initial queries ----
    for (int li = 0; li < NLAY; ++li) {
        cvt(LP(li, 6),  wca_in[li],  768 * 256);
        cvt(LP(li, 8),  wca_out[li], 256 * 256);
        cvt(LP(li, 10), wsa_in[li],  768 * 256);
        cvt(LP(li, 12), wsa_out[li], 256 * 256);
        cvt(LP(li, 14), wff1[li],    1024 * 256);
        cvt(LP(li, 16), wff2[li],    256 * 1024);
    }
    k_pixT<<<dim3(PIXHW / 32, DIMC / 32, NBATCH), dim3(32, 8), 0, stream>>>(pix, pixT);
    k_ln<<<dim3((NBATCH * NGLOB + 7) / 8), 256, 0, stream>>>(gtok, p_gtg, p_gtb,
                                                             nullptr, gtln, NBATCH * NGLOB);
    for (int l = 0; l < 3; ++l) {
        int HW = Hs[l] * Hs[l];
        k_mem<<<dim3(HW / 32, DIMC / 32, NBATCH), dim3(32, 8), 0, stream>>>(
            feat[l], memL[l], Hs[l], Hs[l], Ss[l]);
        k_gtmem<<<dim3((NBATCH * NGLOB * DIMC + 255) / 256), 256, 0, stream>>>(gtln, memL[l], Ss[l]);
    }
    k_initq<<<dim3((nXC + 255) / 256), 256, 0, stream>>>(p_q, X, Xb);

    // ---- predict helper ----
    auto predict = [&](float* auxDst, float* mlDst) {
        gemm(stream, Xb, pixT, nullptr, pqm, nullptr, NQ, PIXHW, DIMC, DIMC, DIMC, PIXHW,
             (long)NQ * DIMC, 0, (long)PIXHW * DIMC, 0, (long)NQ * PIXHW, 0, 1, NBATCH, 1.f);
        k_cls<<<dim3((NBATCH * NQ + 7) / 8), 256, 0, stream>>>(X, p_cw, p_cb, clsb, out_cs);
        k_softq<<<dim3(NBATCH), 128, 0, stream>>>(clsb, wq);
        k_ml<<<dim3(NBATCH * PIXHW / 256), 256, 0, stream>>>(pqm, wq, auxDst, mlDst);
    };
    predict(out_ax, nullptr);

    // ---- decoder layers ----
    for (int li = 0; li < NLAY; ++li) {
        int lvl = li % 3;
        int S = Ss[lvl], Sp = Sps[lvl], H = Hs[lvl];

        // cross-attention mask from current pqm
        k_mask<<<dim3(NBATCH * NQ), 256, 0, stream>>>(pqm, maskb, H, H, S, Sp);
        // x = q + qpos
        k_addpos<<<dim3((nXC + 255) / 256), 256, 0, stream>>>(X, p_qpos);

        // ---- cross-attention ----
        k_ln<<<dim3((NBATCH * NQ + 7) / 8), 256, 0, stream>>>(X, LP(li, 0), LP(li, 1),
                                                              lnb, nullptr, NBATCH * NQ);
        gemm(stream, lnb, wca_in[li], LP(li, 7), nullptr, qpb,
             NBATCH * NQ, DIMC, DIMC, DIMC, DIMC, DIMC, 0, 0, 0, 0, 0, 0, 1, 1, 1.f);
        gemm(stream, memL[lvl], wca_in[li] + DIMC * DIMC, LP(li, 7) + DIMC, nullptr, kvb,
             NBATCH * S, 2 * DIMC, DIMC, DIMC, DIMC, 512, 0, 0, 0, 0, 0, 0, 1, 1, 1.f);
        k_vT<<<dim3((NBATCH * NHD * DHEAD * Sp + 255) / 256), 256, 0, stream>>>(
            kvb, vT, 512, 256, S, Sp);
        gemm(stream, qpb, kvb, nullptr, logit, nullptr, NQ, Sp, DHEAD, DIMC, 512, Sp,
             (long)NQ * DIMC, DHEAD, (long)S * 512, DHEAD,
             (long)NHD * NQ * Sp, (long)NQ * Sp, NHD, NBATCH, ascale);
        k_softmax<<<dim3(NBATCH * NHD * NQ), 256, 0, stream>>>(logit, maskb, probs, S, Sp, NHD);
        gemm(stream, probs, vT, nullptr, nullptr, ob, NQ, DHEAD, Sp, Sp, Sp, DIMC,
             (long)NHD * NQ * Sp, (long)NQ * Sp, (long)NHD * DHEAD * Sp, (long)DHEAD * Sp,
             (long)NQ * DIMC, DHEAD, NHD, NBATCH, 1.f);
        gemm(stream, ob, wca_out[li], LP(li, 9), attn, nullptr,
             NBATCH * NQ, DIMC, DIMC, DIMC, DIMC, DIMC, 0, 0, 0, 0, 0, 0, 1, 1, 1.f);
        k_resadd<<<dim3((nXC + 255) / 256), 256, 0, stream>>>(X, attn, nXC);

        // ---- self-attention ----
        k_ln<<<dim3((NBATCH * NQ + 7) / 8), 256, 0, stream>>>(X, LP(li, 2), LP(li, 3),
                                                              lnb, nullptr, NBATCH * NQ);
        gemm(stream, lnb, wsa_in[li], LP(li, 11), nullptr, qkvb,
             NBATCH * NQ, 3 * DIMC, DIMC, DIMC, DIMC, 768, 0, 0, 0, 0, 0, 0, 1, 1, 1.f);
        k_vT<<<dim3((NBATCH * NHD * DHEAD * 128 + 255) / 256), 256, 0, stream>>>(
            qkvb, vT, 768, 512, NQ, 128);
        gemm(stream, qkvb, qkvb + DIMC, nullptr, logit, nullptr, NQ, 128, DHEAD, 768, 768, 128,
             (long)NQ * 768, DHEAD, (long)NQ * 768, DHEAD,
             (long)NHD * NQ * 128, (long)NQ * 128, NHD, NBATCH, ascale);
        k_softmax<<<dim3(NBATCH * NHD * NQ), 256, 0, stream>>>(logit, nullptr, probs, NQ, 128, NHD);
        gemm(stream, probs, vT, nullptr, nullptr, ob, NQ, DHEAD, 128, 128, 128, DIMC,
             (long)NHD * NQ * 128, (long)NQ * 128, (long)NHD * DHEAD * 128, (long)DHEAD * 128,
             (long)NQ * DIMC, DHEAD, NHD, NBATCH, 1.f);
        gemm(stream, ob, wsa_out[li], LP(li, 13), attn, nullptr,
             NBATCH * NQ, DIMC, DIMC, DIMC, DIMC, DIMC, 0, 0, 0, 0, 0, 0, 1, 1, 1.f);
        k_resadd<<<dim3((nXC + 255) / 256), 256, 0, stream>>>(X, attn, nXC);

        // ---- FFN ----
        k_ln<<<dim3((NBATCH * NQ + 7) / 8), 256, 0, stream>>>(X, LP(li, 4), LP(li, 5),
                                                              lnb, nullptr, NBATCH * NQ);
        gemm(stream, lnb, wff1[li], LP(li, 15), h1, nullptr,
             NBATCH * NQ, 1024, DIMC, DIMC, DIMC, 1024, 0, 0, 0, 0, 0, 0, 1, 1, 1.f);
        k_gelu<<<dim3((NBATCH * NQ * 1024 + 255) / 256), 256, 0, stream>>>(h1, gb, NBATCH * NQ * 1024);
        gemm(stream, gb, wff2[li], LP(li, 17), attn, nullptr,
             NBATCH * NQ, DIMC, 1024, 1024, 1024, DIMC, 0, 0, 0, 0, 0, 0, 1, 1, 1.f);
        k_resadd<<<dim3((nXC + 255) / 256), 256, 0, stream>>>(X, attn, nXC);
        cvt(X, Xb, nXC);

        // ---- predict (aux[li+1]; final layer also writes ml) ----
        predict(out_ax + (size_t)(li + 1) * NBATCH * PIXHW,
                (li == NLAY - 1) ? out_ml : nullptr);
    }
}
